// ForwardDeformer_65463891526389
// MI455X (gfx1250) — compile-verified, matrix operand order checked by reading, not run on hardware
//
#include <hip/hip_runtime.h>
#include <hip/hip_bf16.h>
#include <math.h>

// ---------------- problem constants (from reference) ----------------
#define NJ 24           // joints in voxel / tfs
#define VD 16           // voxel depth  (z)
#define VH 64           // voxel height (y)
#define VW 64           // voxel width  (x)
#define JSTR (VD*VH*VW) // per-joint voxel stride
#define NB 9            // INIT_BONES count
#define SBLEND 20.0f
#define HID 128
#define DLAT 64
#define KIN 67          // 3 + 64
#define LSTR 132        // padded LDS row stride (floats) -> conflict-free column reads

__constant__ int kBones[NB] = {0, 1, 2, 4, 5, 16, 17, 18, 19};

// ---------------- WMMA types / path selection ----------------
typedef __attribute__((ext_vector_type(2)))  float    v2f;
typedef __attribute__((ext_vector_type(8)))  float    v8f;
typedef __attribute__((ext_vector_type(16))) _Float16 v16h;

#if defined(__has_builtin)
#  if __has_builtin(__builtin_amdgcn_wmma_f32_16x16x4_f32)
#    define USE_WMMA_F32 1
#  else
#    define USE_WMMA_F32 0
#  endif
#else
#  define USE_WMMA_F32 0
#endif

// ---------------- trilinear sampling helpers ----------------
struct Corners { float c000,c001,c010,c011,c100,c101,c110,c111; };

__device__ __forceinline__ Corners ldc(const float* __restrict__ vj, int ix, int iy, int iz) {
  const float* p = vj + iz*(VH*VW) + iy*VW + ix;
  Corners c;
  c.c000 = p[0];    c.c001 = p[1];
  c.c010 = p[VW];   c.c011 = p[VW+1];
  p += VH*VW;
  c.c100 = p[0];    c.c101 = p[1];
  c.c110 = p[VW];   c.c111 = p[VW+1];
  return c;
}

__device__ __forceinline__ float tri_val(const Corners& c, float wx, float wy, float wz) {
  float c00 = c.c000 + wx*(c.c001 - c.c000);
  float c01 = c.c010 + wx*(c.c011 - c.c010);
  float c10 = c.c100 + wx*(c.c101 - c.c100);
  float c11 = c.c110 + wx*(c.c111 - c.c110);
  float c0  = c00 + wy*(c01 - c00);
  float c1  = c10 + wy*(c11 - c10);
  return c0 + wz*(c1 - c0);
}

__device__ __forceinline__ void tri_grad(const Corners& c, float wx, float wy, float wz,
                                         float& dwx, float& dwy, float& dwz) {
  float gx00 = c.c001 - c.c000, gx01 = c.c011 - c.c010;
  float gx10 = c.c101 - c.c100, gx11 = c.c111 - c.c110;
  dwx = (gx00 + wy*(gx01 - gx00))*(1.f - wz) + (gx10 + wy*(gx11 - gx10))*wz;
  float c00 = c.c000 + wx*(c.c001 - c.c000);
  float c01 = c.c010 + wx*(c.c011 - c.c010);
  float c10 = c.c100 + wx*(c.c101 - c.c100);
  float c11 = c.c110 + wx*(c.c111 - c.c110);
  dwy = (c01 - c00)*(1.f - wz) + (c11 - c10)*wz;
  float c0 = c00 + wy*(c01 - c00);
  float c1 = c10 + wy*(c11 - c10);
  dwz = c1 - c0;
}

// prep: grid coord + frac + clip-aware derivative (matches jnp.clip JVP)
__device__ __forceinline__ void prep1(float c, int res, int& i0, float& w, float& dg) {
  float scale = 0.5f*(float)(res - 1);
  float hi = (float)(res - 1);
  float g  = (c + 1.0f)*scale;
  float gc = fminf(fmaxf(g, 0.0f), hi);
  dg = (g > 0.0f && g < hi) ? scale : 0.0f;
  int i = (int)floorf(gc);
  if (i > res - 2) i = res - 2;
  if (i < 0) i = 0;
  i0 = i;
  w  = gc - (float)i;
}

// ---------------- skin(x): softmax-blended transform applied to x ----------------
__device__ __forceinline__ void skin_eval(const float* __restrict__ voxel,
                                          const float* __restrict__ tfsb,
                                          float px, float py, float pz, float out[3]) {
  int ix, iy, iz; float wx, wy, wz, dgx, dgy, dgz;
  prep1(px, VW, ix, wx, dgx);
  prep1(py, VH, iy, wy, dgy);
  prep1(pz, VD, iz, wz, dgz);
  float v[NJ]; float m = -3.0e38f;
#pragma unroll
  for (int j = 0; j < NJ; ++j) {
    Corners c = ldc(voxel + j*JSTR, ix, iy, iz);
    v[j] = tri_val(c, wx, wy, wz);
    m = fmaxf(m, v[j]);
  }
  float S = 0.f, T[12];
#pragma unroll
  for (int r = 0; r < 12; ++r) T[r] = 0.f;
#pragma unroll
  for (int j = 0; j < NJ; ++j) {
    float e = expf(SBLEND*(v[j] - m));
    S += e;
    const float* A = tfsb + j*16;
#pragma unroll
    for (int r = 0; r < 12; ++r) T[r] += e*A[r];
  }
  float inv = 1.f/S;
  out[0] = (T[0]*px + T[1]*py + T[2]*pz  + T[3])*inv;
  out[1] = (T[4]*px + T[5]*py + T[6]*pz  + T[7])*inv;
  out[2] = (T[8]*px + T[9]*py + T[10]*pz + T[11])*inv;
}

// ---------------- analytic skin Jacobian (incl. weight-field term) + value ----------------
__device__ __forceinline__ void skin_jacinv_val(const float* __restrict__ voxel,
                                                const float* __restrict__ tfsb,
                                                float px, float py, float pz,
                                                float Ji[9], float sv[3]) {
  int ix, iy, iz; float wx, wy, wz, dgx, dgy, dgz;
  prep1(px, VW, ix, wx, dgx);
  prep1(py, VH, iy, wy, dgy);
  prep1(pz, VD, iz, wz, dgz);
  float v[NJ]; float m = -3.0e38f;
#pragma unroll
  for (int j = 0; j < NJ; ++j) {
    Corners c = ldc(voxel + j*JSTR, ix, iy, iz);
    v[j] = tri_val(c, wx, wy, wz);
    m = fmaxf(m, v[j]);
  }
  float S = 0.f;
  float M1[9], P1[3], G[3], Q[9];
#pragma unroll
  for (int r = 0; r < 9; ++r) { M1[r] = 0.f; Q[r] = 0.f; }
#pragma unroll
  for (int r = 0; r < 3; ++r) { P1[r] = 0.f; G[r] = 0.f; }
#pragma unroll
  for (int j = 0; j < NJ; ++j) {
    Corners c = ldc(voxel + j*JSTR, ix, iy, iz);
    float dwx, dwy, dwz; tri_grad(c, wx, wy, wz, dwx, dwy, dwz);
    float e = expf(SBLEND*(v[j] - m));
    S += e;
    const float* A = tfsb + j*16;
    float y0 = A[0]*px + A[1]*py + A[2]*pz  + A[3];
    float y1 = A[4]*px + A[5]*py + A[6]*pz  + A[7];
    float y2 = A[8]*px + A[9]*py + A[10]*pz + A[11];
    float d0 = dwx*dgx, d1 = dwy*dgy, d2 = dwz*dgz;
    M1[0] += e*A[0]; M1[1] += e*A[1]; M1[2] += e*A[2];
    M1[3] += e*A[4]; M1[4] += e*A[5]; M1[5] += e*A[6];
    M1[6] += e*A[8]; M1[7] += e*A[9]; M1[8] += e*A[10];
    P1[0] += e*y0;   P1[1] += e*y1;   P1[2] += e*y2;
    G[0]  += e*d0;   G[1]  += e*d1;   G[2]  += e*d2;
    Q[0] += e*y0*d0; Q[1] += e*y0*d1; Q[2] += e*y0*d2;
    Q[3] += e*y1*d0; Q[4] += e*y1*d1; Q[5] += e*y1*d2;
    Q[6] += e*y2*d0; Q[7] += e*y2*d1; Q[8] += e*y2*d2;
  }
  float inv = 1.f/S;
  float yb0 = P1[0]*inv, yb1 = P1[1]*inv, yb2 = P1[2]*inv;
  float g0 = G[0]*inv, g1 = G[1]*inv, g2 = G[2]*inv;
  sv[0] = yb0; sv[1] = yb1; sv[2] = yb2;
  float J[9];
  J[0] = M1[0]*inv + SBLEND*(Q[0]*inv - yb0*g0);
  J[1] = M1[1]*inv + SBLEND*(Q[1]*inv - yb0*g1);
  J[2] = M1[2]*inv + SBLEND*(Q[2]*inv - yb0*g2);
  J[3] = M1[3]*inv + SBLEND*(Q[3]*inv - yb1*g0);
  J[4] = M1[4]*inv + SBLEND*(Q[4]*inv - yb1*g1);
  J[5] = M1[5]*inv + SBLEND*(Q[5]*inv - yb1*g2);
  J[6] = M1[6]*inv + SBLEND*(Q[6]*inv - yb2*g0);
  J[7] = M1[7]*inv + SBLEND*(Q[7]*inv - yb2*g1);
  J[8] = M1[8]*inv + SBLEND*(Q[8]*inv - yb2*g2);
  float det = J[0]*(J[4]*J[8]-J[5]*J[7]) - J[1]*(J[3]*J[8]-J[5]*J[6]) + J[2]*(J[3]*J[7]-J[4]*J[6]);
  float id = 1.f/det;
  Ji[0] = (J[4]*J[8]-J[5]*J[7])*id; Ji[1] = (J[2]*J[7]-J[1]*J[8])*id; Ji[2] = (J[1]*J[5]-J[2]*J[4])*id;
  Ji[3] = (J[5]*J[6]-J[3]*J[8])*id; Ji[4] = (J[0]*J[8]-J[2]*J[6])*id; Ji[5] = (J[2]*J[3]-J[0]*J[5])*id;
  Ji[6] = (J[3]*J[7]-J[4]*J[6])*id; Ji[7] = (J[1]*J[6]-J[0]*J[7])*id; Ji[8] = (J[0]*J[4]-J[1]*J[3])*id;
}

// ---------------- per-point: Tinv init + Broyden root search ----------------
__global__ void __launch_bounds__(256)
broyden_kernel(const float* __restrict__ xd, const float* __restrict__ tfs,
               const float* __restrict__ voxel,
               float* __restrict__ xc_out, float* __restrict__ valid_out,
               int B, int N) {
  long NI = (long)N * NB;
  long P  = (long)B * NI;
  long p  = (long)blockIdx.x * blockDim.x + threadIdx.x;
  if (p >= P) return;
  int  b = (int)(p / NI);
  long r = p - (long)b * NI;
  int  n = (int)(r / NB);
  int  i = (int)(r - (long)n * NB);

  const float* xdp = xd + ((long)b * N + n) * 3;
  float tx = xdp[0], ty = xdp[1], tz = xdp[2];

  // affine inverse of tfs[b, INIT_BONES[i]]  (bottom row is [0,0,0,1])
  const float* Mm = tfs + ((long)b * NJ + kBones[i]) * 16;
  float a00 = Mm[0], a01 = Mm[1], a02 = Mm[2],  t0 = Mm[3];
  float a10 = Mm[4], a11 = Mm[5], a12 = Mm[6],  t1 = Mm[7];
  float a20 = Mm[8], a21 = Mm[9], a22 = Mm[10], t2 = Mm[11];
  float det = a00*(a11*a22-a12*a21) - a01*(a10*a22-a12*a20) + a02*(a10*a21-a11*a20);
  float id  = 1.f/det;
  float r00 = (a11*a22-a12*a21)*id, r01 = (a02*a21-a01*a22)*id, r02 = (a01*a12-a02*a11)*id;
  float r10 = (a12*a20-a10*a22)*id, r11 = (a00*a22-a02*a20)*id, r12 = (a02*a10-a00*a12)*id;
  float r20 = (a10*a21-a11*a20)*id, r21 = (a01*a20-a00*a21)*id, r22 = (a00*a11-a01*a10)*id;
  float x0 = r00*tx + r01*ty + r02*tz - (r00*t0 + r01*t1 + r02*t2);
  float x1 = r10*tx + r11*ty + r12*tz - (r10*t0 + r11*t1 + r12*t2);
  float x2 = r20*tx + r21*ty + r22*tz - (r20*t0 + r21*t1 + r22*t2);

  const float* tfsb = tfs + (long)b * NJ * 16;
  float Ji[9], sv[3];
  skin_jacinv_val(voxel, tfsb, x0, x1, x2, Ji, sv);
  float f0 = sv[0] - tx, f1 = sv[1] - ty, f2 = sv[2] - tz;

#pragma unroll 1
  for (int step = 0; step < 5; ++step) {
    float dx0 = -(Ji[0]*f0 + Ji[1]*f1 + Ji[2]*f2);
    float dx1 = -(Ji[3]*f0 + Ji[4]*f1 + Ji[5]*f2);
    float dx2 = -(Ji[6]*f0 + Ji[7]*f1 + Ji[8]*f2);
    x0 += dx0; x1 += dx1; x2 += dx2;
    float s[3]; skin_eval(voxel, tfsb, x0, x1, x2, s);
    float fn0 = s[0] - tx, fn1 = s[1] - ty, fn2 = s[2] - tz;
    float df0 = fn0 - f0, df1 = fn1 - f1, df2 = fn2 - f2;
    float Jd0 = Ji[0]*df0 + Ji[1]*df1 + Ji[2]*df2;
    float Jd1 = Ji[3]*df0 + Ji[4]*df1 + Ji[5]*df2;
    float Jd2 = Ji[6]*df0 + Ji[7]*df1 + Ji[8]*df2;
    float denom = dx0*Jd0 + dx1*Jd1 + dx2*Jd2;
    denom = (fabsf(denom) < 1e-8f) ? 1e-8f : denom;
    float idn = 1.f/denom;
    float u0 = dx0 - Jd0, u1 = dx1 - Jd1, u2 = dx2 - Jd2;
    float vr0 = dx0*Ji[0] + dx1*Ji[3] + dx2*Ji[6];
    float vr1 = dx0*Ji[1] + dx1*Ji[4] + dx2*Ji[7];
    float vr2 = dx0*Ji[2] + dx1*Ji[5] + dx2*Ji[8];
    Ji[0] += u0*vr0*idn; Ji[1] += u0*vr1*idn; Ji[2] += u0*vr2*idn;
    Ji[3] += u1*vr0*idn; Ji[4] += u1*vr1*idn; Ji[5] += u1*vr2*idn;
    Ji[6] += u2*vr0*idn; Ji[7] += u2*vr1*idn; Ji[8] += u2*vr2*idn;
    f0 = fn0; f1 = fn1; f2 = fn2;
  }

  xc_out[p*3 + 0] = x0;
  xc_out[p*3 + 1] = x1;
  xc_out[p*3 + 2] = x2;
  valid_out[p] = (sqrtf(f0*f0 + f1*f1 + f2*f2) < 1e-4f) ? 1.0f : 0.0f;
}

// ---------------- WMMA 16x16 output-tile GEMM helper ----------------
// Computes one 16(M) x 16(N) tile: C += X[16 x Kpad] * W[Kvalid x ncols] column 'col'.
__device__ __forceinline__ v8f mm_tile(const float* __restrict__ Xb, int lane,
                                       const float* __restrict__ Wg,
                                       int ldw, int ncols, int col,
                                       int Kpad, int Kvalid, v8f acc) {
  int mrow = lane & 15;
#if USE_WMMA_F32
  int khalf = (lane >> 4) << 1;                 // lanes 0-15: K {0,1}; 16-31: K {2,3}
  for (int kc = 0; kc < Kpad; kc += 4) {
    int k0 = kc + khalf;
    v2f a; a.x = Xb[mrow*LSTR + k0]; a.y = Xb[mrow*LSTR + k0 + 1];
    int ka = (k0     < Kvalid) ? k0     : (Kvalid - 1);
    int kb = (k0 + 1 < Kvalid) ? k0 + 1 : (Kvalid - 1);
    float w0 = 0.f, w1 = 0.f;
    if (col < ncols) { w0 = Wg[ka*ldw + col]; w1 = Wg[kb*ldw + col]; }
    v2f bb; bb.x = w0; bb.y = w1;
    acc = __builtin_amdgcn_wmma_f32_16x16x4_f32(false, a, false, bb,
                                                (short)0, acc, false, false);
  }
#else
  int kq = (lane >> 4) << 3;                    // lanes 0-15: K base 0; 16-31: K base 8
  int Kp32 = (Kpad + 31) & ~31;
  for (int kc = 0; kc < Kp32; kc += 32) {
    v16h a, bb;
#pragma unroll
    for (int h = 0; h < 16; ++h) {
      int k = kc + kq + (h & 7) + ((h & 8) ? 16 : 0);
      a[h] = (_Float16)Xb[mrow*LSTR + k];
      float wv = 0.f;
      if (col < ncols && k < Kvalid) wv = Wg[k*ldw + col];
      bb[h] = (_Float16)wv;
    }
    acc = __builtin_amdgcn_wmma_f32_16x16x32_f16(false, a, false, bb,
                                                 (short)0, acc, false, false);
  }
#endif
  return acc;
}

__device__ __forceinline__ float softplus_f(float x) {
  return fmaxf(x, 0.f) + log1pf(expf(-fabsf(x)));   // logaddexp(x, 0)
}

// ---------------- MLP: [xc | latent] -> 128 -> 128 -> 3, residual add ----------------
// One wave per 16-point tile; per-wave LDS (no cross-wave barriers needed).
// xc_in aliases 'out' (in-place): each location is read then written by its own wave only.
__global__ void __launch_bounds__(64)
mlp_kernel(const float* __restrict__ xc_in, const float* __restrict__ latent,
           const float* __restrict__ W1, const float* __restrict__ b1,
           const float* __restrict__ W2, const float* __restrict__ b2,
           const float* __restrict__ W3, const float* __restrict__ b3,
           float* __restrict__ out, int B, int N) {
  __shared__ float lds[2][2][16*LSTR];          // [wave][buffer X/H][16 x 132]
  int wave = threadIdx.x >> 5;
  int lane = threadIdx.x & 31;
  long NI = (long)N * NB;
  long P  = (long)B * NI;
  long p0 = ((long)blockIdx.x * 2 + wave) * 16;
  if (p0 >= P) return;
  int b = (int)(p0 / NI);

  float* X = &lds[wave][0][0];
  float* H = &lds[wave][1][0];
  int mrow  = lane & 15;
  int crow0 = (lane >> 4) << 3;                 // C/D fragment row base: 0 or 8

  // Stage input X = [xc(3) | latent(64) | zero pad to 96]
  for (int idx = lane; idx < 16*96; idx += 32) {
    int rr = idx / 96, cc = idx - rr*96;
    float val;
    if (cc < 3)        val = xc_in[(p0 + rr)*3 + cc];
    else if (cc < KIN) val = latent[(long)b*DLAT + (cc - 3)];
    else               val = 0.f;
    X[rr*LSTR + cc] = val;
  }
  __builtin_amdgcn_wave_barrier();

  // Layer 1: X[16x67] @ W1[67x128] + b1 -> softplus -> H
#pragma unroll
  for (int nt = 0; nt < 8; ++nt) {
    v8f acc = {0.f,0.f,0.f,0.f,0.f,0.f,0.f,0.f};
    int col = nt*16 + mrow;
    acc = mm_tile(X, lane, W1, HID, HID, col, 68, KIN, acc);
    float bias = b1[col];
#pragma unroll
    for (int vv = 0; vv < 8; ++vv)
      H[(vv + crow0)*LSTR + col] = softplus_f(acc[vv] + bias);
  }
  __builtin_amdgcn_wave_barrier();

  // Layer 2: H[16x128] @ W2[128x128] + b2 -> softplus -> X (reuse)
#pragma unroll
  for (int nt = 0; nt < 8; ++nt) {
    v8f acc = {0.f,0.f,0.f,0.f,0.f,0.f,0.f,0.f};
    int col = nt*16 + mrow;
    acc = mm_tile(H, lane, W2, HID, HID, col, 128, 128, acc);
    float bias = b2[col];
#pragma unroll
    for (int vv = 0; vv < 8; ++vv)
      X[(vv + crow0)*LSTR + col] = softplus_f(acc[vv] + bias);
  }
  __builtin_amdgcn_wave_barrier();

  // Layer 3: X[16x128] @ W3[128x3] + b3, residual add, store
  {
    v8f acc = {0.f,0.f,0.f,0.f,0.f,0.f,0.f,0.f};
    acc = mm_tile(X, lane, W3, 3, 3, mrow, 128, 128, acc);
    if (mrow < 3) {
      float bias = b3[mrow];
#pragma unroll
      for (int vv = 0; vv < 8; ++vv) {
        long pp = p0 + vv + crow0;
        out[pp*3 + mrow] = acc[vv] + bias + xc_in[pp*3 + mrow];
      }
    }
  }
}

// ---------------- launch ----------------
extern "C" void kernel_launch(void* const* d_in, const int* in_sizes, int n_in,
                              void* d_out, int out_size, void* d_ws, size_t ws_size,
                              hipStream_t stream) {
  const float* xd     = (const float*)d_in[0];
  const float* tfs    = (const float*)d_in[1];
  const float* latent = (const float*)d_in[2];
  const float* voxel  = (const float*)d_in[3];
  const float* W1     = (const float*)d_in[4];
  const float* b1     = (const float*)d_in[5];
  const float* W2     = (const float*)d_in[6];
  const float* b2     = (const float*)d_in[7];
  const float* W3     = (const float*)d_in[8];
  const float* b3     = (const float*)d_in[9];
  (void)n_in; (void)d_ws; (void)ws_size; (void)out_size;

  int B = in_sizes[2] / DLAT;           // latent: [B, 64]
  int N = in_sizes[0] / (3 * B);        // xd: [B, N, 3]
  long P = (long)B * N * NB;

  float* out_xc    = (float*)d_out;     // [P,3] — staged xc_opt, then final xc (in place)
  float* out_valid = out_xc + P * 3;    // [P]

  int bblocks = (int)((P + 255) / 256);
  broyden_kernel<<<bblocks, 256, 0, stream>>>(xd, tfs, voxel, out_xc, out_valid, B, N);

  long tiles = P / 16;                  // P is a multiple of 16 (N*9*2, N=16384)
  int mblocks = (int)((tiles + 1) / 2); // 2 waves (tiles) per 64-thread block
  mlp_kernel<<<mblocks, 64, 0, stream>>>(out_xc, latent, W1, b1, W2, b2, W3, b3,
                                         out_xc, B, N);
}
